// HomoGAT_19121194401862
// MI455X (gfx1250) — compile-verified, matrix operand order checked by reading, not run on hardware
//
#include <hip/hip_runtime.h>
#include <hip/hip_bf16.h>
#include <math.h>

#define IN_F   256
#define DM     256
#define HIDC   64
#define NHEAD  4
#define NGRAPH 64
#define NOUT   2
#define NEG_SLOPE 0.2f

typedef __attribute__((ext_vector_type(16))) __bf16 v16bf;
typedef __attribute__((ext_vector_type(8)))  float  v8f;

union BF16Frag { v16bf v; unsigned short u[16]; };

__device__ __forceinline__ unsigned short f2bf(float f) {
    unsigned int u = __float_as_uint(f);
    unsigned int r = u + 0x7fffu + ((u >> 16) & 1u);   // round-to-nearest-even
    return (unsigned short)(r >> 16);
}

// order-preserving float<->int key for atomicMax on floats
__device__ __forceinline__ int fkey(float f) {
    int i = __float_as_int(f);
    return i >= 0 ? i : (i ^ 0x7fffffff);
}
__device__ __forceinline__ float funkey(int i) {
    return __int_as_float(i >= 0 ? i : (i ^ 0x7fffffff));
}

// ---------------------------------------------------------------------------
// Fragment layout (both A and B packs), per 32-wide K block kb:
//   slot [half*16 + i] holds K = kb + (i<8 ? half*8+i : 16 + half*8 + (i-8))
// matching the ISA 16-bit matrix VGPR layout (half 0: K 0-7,16-23;
// half 1: K 8-15,24-31).
// ---------------------------------------------------------------------------

// Pack W [256(k) x 256(n)] fp32 -> bf16: Wp[((kb/32)*256 + col)*32 + slot]
__global__ void pack_w_kernel(const float* __restrict__ W,
                              unsigned short* __restrict__ Wp) {
    int id = blockIdx.x * blockDim.x + threadIdx.x;
    if (id >= (DM / 32) * DM * 32) return;
    int j   = id & 31;
    int col = (id >> 5) & 255;
    int kbi = id >> 13;
    int kb  = kbi * 32;
    int k;
    if      (j < 8)  k = kb + j;
    else if (j < 16) k = kb + j + 8;
    else if (j < 24) k = kb + j - 8;
    else             k = kb + j;
    Wp[id] = f2bf(W[k * DM + col]);
}

// Pack A [M x 256] fp32 -> bf16: Ap[row*256 + kbi*32 + slot]
__global__ void pack_a_kernel(const float* __restrict__ A,
                              unsigned short* __restrict__ Ap, size_t total) {
    size_t id = (size_t)blockIdx.x * blockDim.x + threadIdx.x;
    if (id >= total) return;
    size_t row = id >> 8;
    int j    = (int)(id & 255);
    int kbi  = j >> 5;
    int jj   = j & 31;
    int half = jj >> 4;
    int i    = jj & 15;
    int k = kbi * 32 + (i < 8 ? half * 8 + i : 16 + half * 8 + (i - 8));
    Ap[id] = f2bf(A[row * 256 + k]);
}

// ---------------------------------------------------------------------------
// C[M x 256] = Ap (bf16 fragment-major) x Wp (bf16 fragment-major), fp32 out.
// 8 waves/block; wave = 16 rows x 64 cols (4 accumulators), K looped 8x32.
// Inner loop: 5 contiguous 32B loads + 4 v_wmma per K step.
// ---------------------------------------------------------------------------
__global__ void __launch_bounds__(256)
gemm_wmma_kernel(const unsigned short* __restrict__ Ap,
                 const unsigned short* __restrict__ Wp,
                 float* __restrict__ C, int M) {
    const int lane = threadIdx.x & 31;
    const int wave = threadIdx.x >> 5;
    const int half = lane >> 4;
    const int r    = lane & 15;
    const int m0   = blockIdx.x * 32 + (wave >> 2) * 16;
    const int n0   = (wave & 3) * 64;

    int row = m0 + r;
    if (row >= M) row = M - 1;
    const unsigned short* __restrict__ arow = Ap + (size_t)row * DM;

    v8f acc0 = {}, acc1 = {}, acc2 = {}, acc3 = {};

    #pragma unroll
    for (int kbi = 0; kbi < 8; ++kbi) {
        BF16Frag a;
        #pragma unroll
        for (int i = 0; i < 16; ++i) a.u[i] = arow[kbi * 32 + half * 16 + i];

        const unsigned short* __restrict__ wb = Wp + (size_t)kbi * DM * 32;
        BF16Frag b0, b1, b2, b3;
        #pragma unroll
        for (int i = 0; i < 16; ++i) b0.u[i] = wb[(n0 +  0 + r) * 32 + half * 16 + i];
        #pragma unroll
        for (int i = 0; i < 16; ++i) b1.u[i] = wb[(n0 + 16 + r) * 32 + half * 16 + i];
        #pragma unroll
        for (int i = 0; i < 16; ++i) b2.u[i] = wb[(n0 + 32 + r) * 32 + half * 16 + i];
        #pragma unroll
        for (int i = 0; i < 16; ++i) b3.u[i] = wb[(n0 + 48 + r) * 32 + half * 16 + i];

        acc0 = __builtin_amdgcn_wmma_f32_16x16x32_bf16(false, a.v, false, b0.v, (short)0, acc0, false, false);
        acc1 = __builtin_amdgcn_wmma_f32_16x16x32_bf16(false, a.v, false, b1.v, (short)0, acc1, false, false);
        acc2 = __builtin_amdgcn_wmma_f32_16x16x32_bf16(false, a.v, false, b2.v, (short)0, acc2, false, false);
        acc3 = __builtin_amdgcn_wmma_f32_16x16x32_bf16(false, a.v, false, b3.v, (short)0, acc3, false, false);
    }

    #pragma unroll
    for (int j = 0; j < 8; ++j) {
        int m = m0 + half * 8 + j;
        if (m < M) {
            float* crow = C + (size_t)m * DM + n0 + r;
            crow[0]  = acc0[j];
            crow[16] = acc1[j];
            crow[32] = acc2[j];
            crow[48] = acc3[j];
        }
    }
}

// ---------------------------------------------------------------------------
__global__ void attn_score_kernel(const float* __restrict__ H,
                                  const float* __restrict__ att_s,
                                  const float* __restrict__ att_d,
                                  float* __restrict__ asrc,
                                  float* __restrict__ adst, int n) {
    int id = blockIdx.x * blockDim.x + threadIdx.x;
    if (id >= n * NHEAD) return;
    int node = id >> 2, h = id & 3;
    const float* hp = H + (size_t)node * DM + h * HIDC;
    const float* sp = att_s + h * HIDC;
    const float* dp = att_d + h * HIDC;
    float ss = 0.f, sd = 0.f;
    #pragma unroll 8
    for (int c = 0; c < HIDC; ++c) { float v = hp[c]; ss += v * sp[c]; sd += v * dp[c]; }
    asrc[id] = ss; adst[id] = sd;
}

__global__ void fill_f32_kernel(float* p, float v, size_t n) {
    size_t i = (size_t)blockIdx.x * blockDim.x + threadIdx.x;
    if (i < n) p[i] = v;
}
__global__ void fill_i32_kernel(int* p, int v, size_t n) {
    size_t i = (size_t)blockIdx.x * blockDim.x + threadIdx.x;
    if (i < n) p[i] = v;
}

__global__ void edge_logits_kernel(const int* __restrict__ ei, int E, int n,
                                   const float* __restrict__ asrc,
                                   const float* __restrict__ adst,
                                   float* __restrict__ ev, int* __restrict__ mmax) {
    int e = blockIdx.x * blockDim.x + threadIdx.x;
    if (e >= E + n) return;
    int s, d;
    if (e < E) { s = ei[e]; d = ei[E + e]; } else { s = d = e - E; }
    #pragma unroll
    for (int h = 0; h < NHEAD; ++h) {
        float v = asrc[s * NHEAD + h] + adst[d * NHEAD + h];
        v = v > 0.f ? v : NEG_SLOPE * v;
        ev[(size_t)e * NHEAD + h] = v;
        atomicMax(&mmax[d * NHEAD + h], fkey(v));
    }
}

__global__ void edge_exp_kernel(const int* __restrict__ ei, int E, int n,
                                float* __restrict__ ev,
                                const int* __restrict__ mmax,
                                float* __restrict__ denom) {
    int e = blockIdx.x * blockDim.x + threadIdx.x;
    if (e >= E + n) return;
    int d;
    if (e < E) { d = ei[E + e]; } else { d = e - E; }
    #pragma unroll
    for (int h = 0; h < NHEAD; ++h) {
        float m = funkey(mmax[d * NHEAD + h]);
        float w = __expf(ev[(size_t)e * NHEAD + h] - m);
        ev[(size_t)e * NHEAD + h] = w;
        atomicAdd(&denom[d * NHEAD + h], w);
    }
}

// one wave32 per edge streams the 256-float message (L2 resident), b128 loads
__global__ void edge_aggr_kernel(const int* __restrict__ ei, int E, int n,
                                 const float* __restrict__ H,
                                 const float* __restrict__ ev,
                                 const float* __restrict__ denom,
                                 float* __restrict__ agg) {
    int wave = blockIdx.x * (blockDim.x >> 5) + (threadIdx.x >> 5);
    int lane = threadIdx.x & 31;
    if (wave >= E + n) return;
    int e = wave, s, d;
    if (e < E) { s = ei[e]; d = ei[E + e]; } else { s = d = e - E; }
    float alpha[NHEAD];
    #pragma unroll
    for (int h = 0; h < NHEAD; ++h)
        alpha[h] = ev[(size_t)e * NHEAD + h] / denom[d * NHEAD + h];
    const float4* hs4 = (const float4*)(H + (size_t)s * DM);
    float* od = agg + (size_t)d * DM;
    #pragma unroll
    for (int it = 0; it < 2; ++it) {
        int q = lane + it * 32;          // float4 index within the row
        float4 v = hs4[q];
        float al = alpha[q >> 4];        // (q*4)>>6 == q>>4
        int b = q * 4;
        atomicAdd(&od[b + 0], v.x * al);
        atomicAdd(&od[b + 1], v.y * al);
        atomicAdd(&od[b + 2], v.z * al);
        atomicAdd(&od[b + 3], v.w * al);
    }
}

// LayerNorm + ReLU, fp32 output
__global__ void __launch_bounds__(256)
ln_relu_kernel(const float* __restrict__ X, const float* __restrict__ bias,
               const float* __restrict__ g, const float* __restrict__ be,
               float* __restrict__ Y, int n) {
    __shared__ float sred[DM];
    int row = blockIdx.x;
    if (row >= n) return;
    int t = threadIdx.x;
    float v = X[(size_t)row * DM + t] + bias[t];
    sred[t] = v; __syncthreads();
    for (int s2 = DM / 2; s2 > 0; s2 >>= 1) { if (t < s2) sred[t] += sred[t + s2]; __syncthreads(); }
    float mu = sred[0] * (1.0f / DM); __syncthreads();
    float dv = v - mu;
    sred[t] = dv * dv; __syncthreads();
    for (int s2 = DM / 2; s2 > 0; s2 >>= 1) { if (t < s2) sred[t] += sred[t + s2]; __syncthreads(); }
    float var = sred[0] * (1.0f / DM);
    float y = dv * rsqrtf(var + 1e-5f) * g[t] + be[t];
    Y[(size_t)row * DM + t] = fmaxf(y, 0.f);
}

// LayerNorm + ReLU, writing bf16 fragment-major output (feeds next GEMM)
__global__ void __launch_bounds__(256)
ln_relu_pack_kernel(const float* __restrict__ X, const float* __restrict__ bias,
                    const float* __restrict__ g, const float* __restrict__ be,
                    unsigned short* __restrict__ Ap, int n) {
    __shared__ float sred[DM];
    int row = blockIdx.x;
    if (row >= n) return;
    int t = threadIdx.x;
    float v = X[(size_t)row * DM + t] + bias[t];
    sred[t] = v; __syncthreads();
    for (int s2 = DM / 2; s2 > 0; s2 >>= 1) { if (t < s2) sred[t] += sred[t + s2]; __syncthreads(); }
    float mu = sred[0] * (1.0f / DM); __syncthreads();
    float dv = v - mu;
    sred[t] = dv * dv; __syncthreads();
    for (int s2 = DM / 2; s2 > 0; s2 >>= 1) { if (t < s2) sred[t] += sred[t + s2]; __syncthreads(); }
    float var = sred[0] * (1.0f / DM);
    float y = fmaxf(dv * rsqrtf(var + 1e-5f) * g[t] + be[t], 0.f);
    // scatter into fragment-major slot
    int kbi = t >> 5;
    int kk  = t & 31;
    int half, i;
    if      (kk < 8)  { half = 0; i = kk; }
    else if (kk < 16) { half = 1; i = kk - 8; }
    else if (kk < 24) { half = 0; i = 8 + (kk - 16); }
    else              { half = 1; i = 8 + (kk - 24); }
    Ap[(size_t)row * DM + kbi * 32 + half * 16 + i] = f2bf(y);
}

__global__ void pool_sum_kernel(const float* __restrict__ H,
                                const int* __restrict__ batch,
                                float* __restrict__ sums, int n) {
    size_t id = (size_t)blockIdx.x * blockDim.x + threadIdx.x;
    if (id >= (size_t)n * DM) return;
    int node = (int)(id >> 8);
    int c = (int)(id & 255);
    atomicAdd(&sums[(size_t)batch[node] * DM + c], H[id]);
}
__global__ void pool_cnt_kernel(const int* __restrict__ batch,
                                float* __restrict__ cnt, int n) {
    int i = blockIdx.x * blockDim.x + threadIdx.x;
    if (i < n) atomicAdd(&cnt[batch[i]], 1.0f);
}

__global__ void mlp1_kernel(const float* __restrict__ sums, const float* __restrict__ cnt,
                            const float* __restrict__ lw1, const float* __restrict__ lb1,
                            float* __restrict__ hid) {
    int id = blockIdx.x * blockDim.x + threadIdx.x;
    if (id >= NGRAPH * HIDC) return;
    int g = id / HIDC, j = id % HIDC;
    float inv = 1.0f / fmaxf(cnt[g], 1.0f);
    float acc = lb1[j];
    for (int k = 0; k < DM; ++k) acc += sums[g * DM + k] * inv * lw1[k * HIDC + j];
    hid[id] = fmaxf(acc, 0.f);
}
__global__ void mlp2_kernel(const float* __restrict__ hid, const float* __restrict__ lw2,
                            const float* __restrict__ lb2, float* __restrict__ out) {
    int id = blockIdx.x * blockDim.x + threadIdx.x;
    if (id >= NGRAPH * NOUT) return;
    int g = id / NOUT, o = id % NOUT;
    float acc = lb2[o];
    for (int k = 0; k < HIDC; ++k) acc += hid[g * HIDC + k] * lw2[k * NOUT + o];
    out[id] = acc;
}

// ---------------------------------------------------------------------------
extern "C" void kernel_launch(void* const* d_in, const int* in_sizes, int n_in,
                              void* d_out, int out_size, void* d_ws, size_t ws_size,
                              hipStream_t stream) {
    const float* x     = (const float*)d_in[0];
    const int*   ei    = (const int*)d_in[1];
    const int*   batch = (const int*)d_in[2];
    const float* W1    = (const float*)d_in[3];
    const float* as1   = (const float*)d_in[4];
    const float* ad1   = (const float*)d_in[5];
    const float* b1    = (const float*)d_in[6];
    const float* g1    = (const float*)d_in[7];
    const float* be1   = (const float*)d_in[8];
    const float* W2    = (const float*)d_in[9];
    const float* as2   = (const float*)d_in[10];
    const float* ad2   = (const float*)d_in[11];
    const float* b2    = (const float*)d_in[12];
    const float* g2    = (const float*)d_in[13];
    const float* be2   = (const float*)d_in[14];
    const float* lw1   = (const float*)d_in[15];
    const float* lb1   = (const float*)d_in[16];
    const float* lw2   = (const float*)d_in[17];
    const float* lb2   = (const float*)d_in[18];
    float* out = (float*)d_out;
    (void)n_in; (void)out_size; (void)ws_size;

    const int N  = in_sizes[0] / IN_F;
    const int E  = in_sizes[1] / 2;
    const int Ep = E + N;

    char* base = (char*)d_ws;
    size_t off = 0;
    auto alloc = [&](size_t bytes) -> char* {
        char* p = base + off;
        off += (bytes + 255) & ~(size_t)255;
        return p;
    };
    unsigned short* Wp1 = (unsigned short*)alloc((size_t)DM * DM * 2);
    unsigned short* Wp2 = (unsigned short*)alloc((size_t)DM * DM * 2);
    unsigned short* Ap  = (unsigned short*)alloc((size_t)N * DM * 2);
    float* buf0  = (float*)alloc((size_t)N * DM * 4);
    float* buf1  = (float*)alloc((size_t)N * DM * 4);
    float* asrc  = (float*)alloc((size_t)N * NHEAD * 4);
    float* adst  = (float*)alloc((size_t)N * NHEAD * 4);
    float* ev    = (float*)alloc((size_t)Ep * NHEAD * 4);
    int*   mmax  = (int*)  alloc((size_t)N * NHEAD * 4);
    float* denom = (float*)alloc((size_t)N * NHEAD * 4);
    float* sums  = (float*)alloc((size_t)NGRAPH * DM * 4);
    float* cnt   = (float*)alloc((size_t)NGRAPH * 4);
    float* hid   = (float*)alloc((size_t)NGRAPH * HIDC * 4);

    const dim3 blk(256);
    const int KEY_NEG_INF = (int)0x807fffff;  // fkey(-inf)
    const int gemmBlocks  = (N + 31) / 32;
    const int nhBlocks    = (N * NHEAD + 255) / 256;
    const int edgeBlocks  = (Ep + 255) / 256;
    const int aggrBlocks  = (Ep + 7) / 8;
    const unsigned ndBlocks = (unsigned)(((size_t)N * DM + 255) / 256);

    pack_w_kernel<<<(DM * DM + 255) / 256, blk, 0, stream>>>(W1, Wp1);
    pack_w_kernel<<<(DM * DM + 255) / 256, blk, 0, stream>>>(W2, Wp2);

    // ---------------- Layer 1 ----------------
    pack_a_kernel<<<ndBlocks, blk, 0, stream>>>(x, Ap, (size_t)N * DM);
    gemm_wmma_kernel<<<gemmBlocks, blk, 0, stream>>>(Ap, Wp1, buf0, N);
    attn_score_kernel<<<nhBlocks, blk, 0, stream>>>(buf0, as1, ad1, asrc, adst, N);
    fill_i32_kernel<<<nhBlocks, blk, 0, stream>>>(mmax, KEY_NEG_INF, (size_t)N * NHEAD);
    fill_f32_kernel<<<nhBlocks, blk, 0, stream>>>(denom, 0.f, (size_t)N * NHEAD);
    edge_logits_kernel<<<edgeBlocks, blk, 0, stream>>>(ei, E, N, asrc, adst, ev, mmax);
    edge_exp_kernel<<<edgeBlocks, blk, 0, stream>>>(ei, E, N, ev, mmax, denom);
    fill_f32_kernel<<<ndBlocks, blk, 0, stream>>>(buf1, 0.f, (size_t)N * DM);
    edge_aggr_kernel<<<aggrBlocks, blk, 0, stream>>>(ei, E, N, buf0, ev, denom, buf1);
    ln_relu_pack_kernel<<<N, blk, 0, stream>>>(buf1, b1, g1, be1, Ap, N);

    // ---------------- Layer 2 ----------------
    gemm_wmma_kernel<<<gemmBlocks, blk, 0, stream>>>(Ap, Wp2, buf0, N);
    attn_score_kernel<<<nhBlocks, blk, 0, stream>>>(buf0, as2, ad2, asrc, adst, N);
    fill_i32_kernel<<<nhBlocks, blk, 0, stream>>>(mmax, KEY_NEG_INF, (size_t)N * NHEAD);
    fill_f32_kernel<<<nhBlocks, blk, 0, stream>>>(denom, 0.f, (size_t)N * NHEAD);
    edge_logits_kernel<<<edgeBlocks, blk, 0, stream>>>(ei, E, N, asrc, adst, ev, mmax);
    edge_exp_kernel<<<edgeBlocks, blk, 0, stream>>>(ei, E, N, ev, mmax, denom);
    fill_f32_kernel<<<ndBlocks, blk, 0, stream>>>(buf1, 0.f, (size_t)N * DM);
    edge_aggr_kernel<<<aggrBlocks, blk, 0, stream>>>(ei, E, N, buf0, ev, denom, buf1);
    ln_relu_kernel<<<N, blk, 0, stream>>>(buf1, b2, g2, be2, buf0, N);

    // ---------------- Pool + MLP ----------------
    fill_f32_kernel<<<(NGRAPH * DM + 255) / 256, blk, 0, stream>>>(sums, 0.f, (size_t)NGRAPH * DM);
    fill_f32_kernel<<<1, blk, 0, stream>>>(cnt, 0.f, (size_t)NGRAPH);
    pool_sum_kernel<<<ndBlocks, blk, 0, stream>>>(buf0, batch, sums, N);
    pool_cnt_kernel<<<(N + 255) / 256, blk, 0, stream>>>(batch, cnt, N);
    mlp1_kernel<<<(NGRAPH * HIDC + 255) / 256, blk, 0, stream>>>(sums, cnt, lw1, lb1, hid);
    mlp2_kernel<<<1, blk, 0, stream>>>(hid, lw2, lb2, out);
}
